// MultiHeadMultiGranAttention_78632261255581
// MI455X (gfx1250) — matrix operand order repaired and checked by reading, not compiled, not run on hardware
//
#include <hip/hip_runtime.h>

// ---------------------------------------------------------------------------
// MI455X (gfx1250) implementation: wave32, WMMA f32_16x16x32_f16 + TDM.
// qkv GEMM (WMMA) -> per-tile attention (TDM-staged K/V^T tiles, WMMA QK^T +
// WMMA PV, relative-PE via distance tables, exact softmax) -> output GEMM.
// ---------------------------------------------------------------------------

typedef _Float16 f16;
typedef __attribute__((ext_vector_type(16))) _Float16 v16h;
typedef __attribute__((ext_vector_type(8)))  float    v8f;

#define NHEADS 16
#define HD     64
#define DIMD   1024
#define BATCH  8
#define STOKL  512
#define STML   128
#define SEQ    640
#define NEG_INF_F (-1e18f)

#if defined(__has_builtin)
#if __has_builtin(__builtin_amdgcn_tensor_load_to_lds) && \
    __has_builtin(__builtin_amdgcn_s_wait_tensorcnt)
#define HAVE_TDM 1
#endif
#endif
#ifndef HAVE_TDM
#define HAVE_TDM 0
#endif

__device__ __forceinline__ v8f wmma_f16(v16h a, v16h b, v8f c) {
  // 8 args: (neg_a, A, neg_b, B, c_mod, C, reuse_a, reuse_b)
  return __builtin_amdgcn_wmma_f32_16x16x32_f16(false, a, false, b, (short)0, c,
                                                false, false);
}

// 16-bit A-matrix 16x32 (MxK) fragment from LDS [16][stride]:
// lanes 0-15: M=lane, K={0..7,16..23}; lanes 16-31: M=lane-16, K={8..15,24..31}
__device__ __forceinline__ v16h load_a_frag_lds(const f16* As, int stride, int lane) {
  const int m = lane & 15, hi = lane >> 4;
  v16h a;
#pragma unroll
  for (int i = 0; i < 8; ++i) {
    const int kb = ((i >> 2) << 4) + (hi << 3) + ((i & 3) << 1);
    a[2 * i]     = As[m * stride + kb];
    a[2 * i + 1] = As[m * stride + kb + 1];
  }
  return a;
}

// 16-bit B-matrix 32x16 (KxN) fragment from LDS stored as [n][kk] (stride):
// lanes 0-15: N=lane, K={0..7,16..23}; lanes 16-31: N=lane-16, K={8..15,24..31}
__device__ __forceinline__ v16h load_b_frag_lds(const f16* Bs, int stride, int lane) {
  const int n = lane & 15, hi = lane >> 4;
  v16h b;
#pragma unroll
  for (int i = 0; i < 8; ++i) {
    const int kb = ((i >> 2) << 4) + (hi << 3) + ((i & 3) << 1);
    b[2 * i]     = Bs[n * stride + kb];
    b[2 * i + 1] = Bs[n * stride + kb + 1];
  }
  return b;
}

#if HAVE_TDM
typedef unsigned int u32x4 __attribute__((ext_vector_type(4)));
typedef int          i32x8 __attribute__((ext_vector_type(8)));
typedef int          i32x4 __attribute__((ext_vector_type(4)));

// Issue one 2D tensor tile load Global->LDS via the Tensor Data Mover.
// Descriptor bit layout per CDNA5 ISA ch.8 (D# group0 128b, group1 256b).
// data_size=1 (2-byte elements). Padding inserts pad_amount DWORDs every
// (pad-interval) so the LDS tile keeps our bank-padded row stride.
__device__ __forceinline__ void tdm_load_2d(unsigned lds_off, const void* gaddr,
                                            unsigned tensor_d0, unsigned tensor_d1,
                                            unsigned tile_d0, unsigned tile_d1,
                                            unsigned stride0_elems,
                                            unsigned pad_interval_code,
                                            unsigned pad_amount_code) {
  const unsigned long long ga = (unsigned long long)(size_t)gaddr;
  u32x4 g0;
  g0[0] = 1u;                                    // count=1, user descriptor
  g0[1] = lds_off;                               // lds_addr (bytes)
  g0[2] = (unsigned)(ga & 0xffffffffu);          // global_addr[31:0]
  g0[3] = (unsigned)((ga >> 32) & 0x01ffffffu)   // global_addr[56:32]
          | 0x80000000u;                         // type=2 ("image")
  i32x8 g1;
  g1[0] = (int)((1u << 16)                       // data_size: 2 bytes
                | (1u << 20)                     // pad_enable
                | (pad_interval_code << 22)
                | (pad_amount_code << 25));
  g1[1] = (int)((tensor_d0 & 0xffffu) << 16);    // tensor_dim0[15:0] @ bits79:48
  g1[2] = (int)(((tensor_d0 >> 16) & 0xffffu) | ((tensor_d1 & 0xffffu) << 16));
  g1[3] = (int)(((tensor_d1 >> 16) & 0xffffu) | ((tile_d0 & 0xffffu) << 16));
  g1[4] = (int)(tile_d1 & 0xffffu);              // tile_dim1; tile_dim2=0
  g1[5] = (int)stride0_elems;                    // tensor_dim0_stride[31:0]
  g1[6] = 0;
  g1[7] = 0;
  i32x4 z4 = {0, 0, 0, 0};
#if __clang_major__ >= 23
  i32x8 z8 = {0, 0, 0, 0, 0, 0, 0, 0};
  __builtin_amdgcn_tensor_load_to_lds(g0, g1, z4, z4, z8, 0);
#else
  __builtin_amdgcn_tensor_load_to_lds(g0, g1, z4, z4, 0);
#endif
}
#endif  // HAVE_TDM

// ---------------------------------------------------------------------------
// Kernel 1: QKV projection. x[B*S,1024] @ qkv_w^T[1024,3072] + b.
// Block = 128 threads (4 waves); block covers 16 rows x 256 cols.
// Outputs: q (pre-scaled 1/8) and k head-major [B,H,S,64]; v TRANSPOSED as
// vT[B,H,64,S] so the attention PV fragments are contiguous-pair loads.
// ---------------------------------------------------------------------------
__global__ __launch_bounds__(128)
void qkv_proj_kernel(const float* __restrict__ stok, const float* __restrict__ stm,
                     const float* __restrict__ qkv_w, const float* __restrict__ qkv_b,
                     f16* __restrict__ qb, f16* __restrict__ kb, f16* __restrict__ vT) {
  __shared__ f16 As[16][36];        // 16x32 + pad
  __shared__ f16 Bs[4][64][34];     // per-wave 64 cols x 32 kk + pad
  const int lane = threadIdx.x & 31;
  const int wave = threadIdx.x >> 5;
  const int tid  = threadIdx.x;
  const int row0 = blockIdx.x * 16;       // global row (b*640+s), tile-aligned
  const int b    = row0 / SEQ;
  const int s0   = row0 % SEQ;            // 16-aligned; never crosses 512 boundary
  const int c0   = blockIdx.y * 256 + wave * 64;

  v8f acc[4] = {};

  for (int k0 = 0; k0 < DIMD; k0 += 32) {
    __syncthreads();
    // stage A tile (16x32): 128 threads x 4 contiguous floats
    {
      const int e0 = tid * 4;
      const int r  = e0 >> 5;
      const int kk = e0 & 31;
      const int s  = s0 + r;
      const float* src = (s < STOKL)
          ? &stok[(size_t)(b * STOKL + s) * DIMD + k0 + kk]
          : &stm [(size_t)(b * STML + (s - STOKL)) * DIMD + k0 + kk];
      if (k0 + 32 < DIMD) __builtin_prefetch(src + 32, 0, 3);
      float4 x4 = *(const float4*)src;
      As[r][kk + 0] = (f16)x4.x; As[r][kk + 1] = (f16)x4.y;
      As[r][kk + 2] = (f16)x4.z; As[r][kk + 3] = (f16)x4.w;
    }
    // stage B tile per wave: B[kk][n] = qkv_w[(c0+n)*1024 + k0+kk], stored n-major
#pragma unroll
    for (int j = 0; j < 2; ++j) {
      const int n = lane + j * 32;
      const float* src = &qkv_w[(size_t)(c0 + n) * DIMD + k0];
#pragma unroll
      for (int q4 = 0; q4 < 8; ++q4) {
        float4 w4 = *(const float4*)(src + q4 * 4);
        Bs[wave][n][q4 * 4 + 0] = (f16)w4.x; Bs[wave][n][q4 * 4 + 1] = (f16)w4.y;
        Bs[wave][n][q4 * 4 + 2] = (f16)w4.z; Bs[wave][n][q4 * 4 + 3] = (f16)w4.w;
      }
    }
    __syncthreads();
    v16h a = load_a_frag_lds(&As[0][0], 36, lane);
#pragma unroll
    for (int nt = 0; nt < 4; ++nt) {
      v16h bfr = load_b_frag_lds(&Bs[wave][nt * 16][0], 34, lane);
      acc[nt] = wmma_f16(a, bfr, acc[nt]);
    }
  }

  // D layout: M = (lane>>4)*8 + vgpr, N = lane&15.
  // cbase is wave-uniform: force it scalar (readfirstlane) so the q/k/v
  // routing becomes s_cbranch instead of per-lane exec masking.
  const int hi = lane >> 4;
  const int nl = lane & 15;
#pragma unroll
  for (int nt = 0; nt < 4; ++nt) {
    const int cbase = __builtin_amdgcn_readfirstlane(c0 + nt * 16);
    const int part  = cbase >> 10;         // 0=q 1=k 2=v
    const int head  = (cbase >> 6) & 15;
    const int hd0   = cbase & 63;          // 0/16/32/48
    const float scale = (part == 0) ? 0.125f : 1.0f;   // 1/sqrt(HEAD_DIM) on q
    const float bias  = qkv_b[cbase + nl];
#pragma unroll
    for (int r = 0; r < 8; ++r) {
      const int m = hi * 8 + r;
      const int s = s0 + m;
      const float v = (acc[nt][r] + bias) * scale;
      const int hdi = hd0 + nl;
      if (part == 2) {
        vT[((size_t)((b * NHEADS + head) * HD + hdi)) * SEQ + s] = (f16)v;
      } else {
        f16* dst = (part == 0) ? qb : kb;
        dst[((size_t)((b * NHEADS + head) * SEQ + s)) * HD + hdi] = (f16)v;
      }
    }
  }
}

// ---------------------------------------------------------------------------
// Kernel 2: attention. One wave per (b, h, 16-query tile). Never materializes
// rel_pe[S,S,hd]: bias(q,k) = Ptab[q][clip(k-q)] with Ptab = q_tile @ emb_k^T,
// and the rel-V term = Wtab @ emb_v with Wtab[q][d] = per-distance prob sums.
// K tiles / V^T tiles land in LDS via the Tensor Data Mover (padded rows keep
// the bank-friendly strides); falls back to coalesced b128 staging.
// ---------------------------------------------------------------------------
__global__ __launch_bounds__(32)
void attn_kernel(const f16* __restrict__ qb, const f16* __restrict__ kb,
                 const f16* __restrict__ vT,
                 const unsigned char* __restrict__ stok_mask,
                 const unsigned char* __restrict__ stm_mask,
                 const float* __restrict__ rel_emb_k, const float* __restrict__ rel_emb_v,
                 const float* __restrict__ stm_emb_k, const float* __restrict__ stm_emb_v,
                 float* __restrict__ attn_out, f16* __restrict__ ctx) {
  __shared__ f16   Qs[16][72];     // q tile 16x64 + pad           (2304 B)
  __shared__ float Ps[16][648];    // scores/probs 16x640 + pad    (41472 B)
  __shared__ float Ptab[16][66];   // rel-k bias table             (4224 B)
  __shared__ float Wtab[16][66];   // rel-v distance prob sums     (4224 B)
  __shared__ f16   Kt[16][72];     // K tile [key][hd], 144B rows  (2304 B)
  __shared__ f16   Vt[64][40];     // V^T tile [hd][key], 80B rows (5120 B)

  const int lane = threadIdx.x;
  const int qt = blockIdx.x % (SEQ / 16);
  const int h  = (blockIdx.x / (SEQ / 16)) % NHEADS;
  const int b  = blockIdx.x / ((SEQ / 16) * NHEADS);
  const int q0 = qt * 16;
  const bool stmRegion = (q0 >= STOKL);      // 512 is 16-aligned: tile is uniform
  const int vocab = stmRegion ? 33 : 65;
  const float* embK = stmRegion ? stm_emb_k : rel_emb_k;
  const float* embV = stmRegion ? stm_emb_v : rel_emb_v;

  const size_t bh = (size_t)(b * NHEADS + h);
  const f16* qbase = qb + bh * SEQ * HD;
  const f16* kbase = kb + bh * SEQ * HD;
  const f16* vbase = vT + bh * HD * SEQ;     // transposed: [hd][seq]

  for (int e = lane; e < 16 * HD; e += 32) {
    const int r = e >> 6, kk = e & 63;
    Qs[r][kk] = qbase[(q0 + r) * HD + kk];
  }
  __syncthreads();

  // Ptab[r][d] = dot(q_row_r (already /8), emb_k[d]); also zero Wtab.
  for (int e = lane; e < 16 * vocab; e += 32) {
    const int r = e / vocab, d = e % vocab;
    float s = 0.f;
    for (int kk = 0; kk < HD; ++kk) s += (float)Qs[r][kk] * embK[d * HD + kk];
    Ptab[r][d] = s;
    Wtab[r][d] = 0.f;
  }

  const int hi = lane >> 4;
  const int nl = lane & 15;

  // ---- scores: S_tile[16x16] = Q[16x64] @ K^T, two K=32 WMMA steps ----
  for (int kt = 0; kt < SEQ / 16; ++kt) {
    const int k0 = kt * 16;
    __syncthreads();   // prior-iteration Kt fragment reads complete (DScnt 0)
#if HAVE_TDM
    // 16 rows x 64 f16: row = 128B (interval code 4), pad 16B (code 3) -> stride 72
    tdm_load_2d((unsigned)(size_t)&Kt[0][0], &kbase[(size_t)k0 * HD],
                HD, (unsigned)(SEQ - k0), HD, 16, HD, 4, 3);
    __builtin_amdgcn_s_wait_tensorcnt(0);
    __syncthreads();
#else
    for (int e = lane; e < 128; e += 32) {
      const int r = e >> 3, c = (e & 7) * 8;
      *(float4*)&Kt[r][c] = *(const float4*)&kbase[(size_t)(k0 + r) * HD + c];
    }
    __syncthreads();
#endif
    if (kt + 1 < SEQ / 16)
      __builtin_prefetch(&kbase[(size_t)(k0 + 16 + (lane >> 1)) * HD + (lane & 1) * 32], 0, 3);

    v8f acc = {};
#pragma unroll
    for (int ks = 0; ks < 2; ++ks) {
      v16h a  = load_a_frag_lds(&Qs[0][ks * 32], 72, lane);
      v16h bf = load_b_frag_lds(&Kt[0][ks * 32], 72, lane);  // B[kk][n]=K[k0+n][kk]
      acc = wmma_f16(a, bf, acc);
    }
    const int kg = k0 + nl;
    const bool pad = (kg < STOKL) ? (stok_mask[b * STOKL + kg] != 0)
                                  : (stm_mask[b * STML + (kg - STOKL)] != 0);
#pragma unroll
    for (int r = 0; r < 8; ++r) {
      const int m  = hi * 8 + r;
      const int qg = q0 + m;
      float v = acc[r];
      if (!stmRegion) {
        if (kg < STOKL) {           // stok->stok: rel bias; stok->stm: zero bias
          int d = kg - qg; d = d < -32 ? -32 : (d > 32 ? 32 : d);
          v += Ptab[m][d + 32];
        }
      } else {
        if (kg >= STOKL) {          // stm->stm: rel bias
          int d = kg - qg; d = d < -16 ? -16 : (d > 16 ? 16 : d);
          v += Ptab[m][d + 16];
        } else {
          v = NEG_INF_F;            // block mask: stm queries can't see stok keys
        }
      }
      if (pad) v = NEG_INF_F;
      Ps[m][kg] = v;
    }
  }
  __syncthreads();

  // ---- softmax per row, emit attn, accumulate distance-bucket sums ----
  float* attn_row_base = attn_out + (bh * SEQ + q0) * SEQ;
  for (int r = 0; r < 16; ++r) {
    const int qg = q0 + r;
    float mx = NEG_INF_F;
    for (int k = lane; k < SEQ; k += 32) mx = fmaxf(mx, Ps[r][k]);
#pragma unroll
    for (int s2 = 16; s2 >= 1; s2 >>= 1) mx = fmaxf(mx, __shfl_xor(mx, s2, 32));
    float sum = 0.f;
    for (int k = lane; k < SEQ; k += 32) {
      const float e = __expf(Ps[r][k] - mx);
      Ps[r][k] = e;
      sum += e;
    }
#pragma unroll
    for (int s2 = 16; s2 >= 1; s2 >>= 1) sum += __shfl_xor(sum, s2, 32);
    const float inv = 1.f / sum;
    for (int k = lane; k < SEQ; k += 32) {
      const float p = Ps[r][k] * inv;
      Ps[r][k] = p;
      attn_row_base[(size_t)r * SEQ + k] = p;
      if (!stmRegion) {
        if (k < STOKL) {
          int d = k - qg; d = d < -32 ? -32 : (d > 32 ? 32 : d);
          atomicAdd(&Wtab[r][d + 32], p);     // ds_add_f32
        }
      } else if (k >= STOKL) {
        int d = k - qg; d = d < -16 ? -16 : (d > 16 ? 16 : d);
        atomicAdd(&Wtab[r][d + 16], p);
      }
    }
  }

  // ---- context = P[16x640] @ V[640x64] via WMMA + Wtab @ emb_v ----
  v8f cacc[4] = {};
  for (int k0 = 0; k0 < SEQ; k0 += 32) {
    __syncthreads();   // prior-iteration Vt fragment reads complete
#if HAVE_TDM
    // 64 rows x 32 f16: row = 64B (interval code 3), pad 16B (code 3) -> stride 40
    tdm_load_2d((unsigned)(size_t)&Vt[0][0], &vbase[k0],
                (unsigned)(SEQ - k0), HD, 32, HD, SEQ, 3, 3);
    __builtin_amdgcn_s_wait_tensorcnt(0);
    __syncthreads();
#else
    for (int e = lane; e < 256; e += 32) {
      const int r = e >> 2, c = (e & 3) * 8;   // r = hd row, c = key chunk
      *(float4*)&Vt[r][c] = *(const float4*)&vbase[(size_t)r * SEQ + k0 + c];
    }
    __syncthreads();
#endif
    if (k0 + 32 < SEQ)
      __builtin_prefetch(&vbase[(size_t)(lane * 2) * SEQ + k0 + 32], 0, 3);

    v16h a;   // A frag from probs (f32 LDS -> f16 at gather), contiguous pairs
#pragma unroll
    for (int i = 0; i < 8; ++i) {
      const int kb = ((i >> 2) << 4) + (hi << 3) + ((i & 3) << 1);
      a[2 * i]     = (f16)Ps[nl][k0 + kb];
      a[2 * i + 1] = (f16)Ps[nl][k0 + kb + 1];
    }
#pragma unroll
    for (int nt = 0; nt < 4; ++nt) {
      v16h bf = load_b_frag_lds(&Vt[nt * 16][0], 40, lane);  // B[kk][n]=V[k][n]
      cacc[nt] = wmma_f16(a, bf, cacc[nt]);
    }
  }

#pragma unroll
  for (int nt = 0; nt < 4; ++nt) {
#pragma unroll
    for (int r = 0; r < 8; ++r) {
      const int m  = hi * 8 + r;
      const int ng = nt * 16 + nl;
      float v = cacc[nt][r];
      for (int d = 0; d < vocab; ++d) v += Wtab[m][d] * embV[d * HD + ng];
      ctx[(size_t)(b * SEQ + q0 + m) * DIMD + h * HD + ng] = (f16)v;
    }
  }
}

// ---------------------------------------------------------------------------
// Kernel 3: output projection. ctx[5120,1024](f16) @ out_w^T + out_b -> fp32.
// ---------------------------------------------------------------------------
__global__ __launch_bounds__(128)
void out_proj_kernel(const f16* __restrict__ ctx, const float* __restrict__ out_w,
                     const float* __restrict__ out_b, float* __restrict__ out) {
  __shared__ f16 As[16][36];
  __shared__ f16 Bs[4][64][34];
  const int lane = threadIdx.x & 31;
  const int wave = threadIdx.x >> 5;
  const int tid  = threadIdx.x;
  const int row0 = blockIdx.x * 16;
  const int c0   = blockIdx.y * 256 + wave * 64;

  v8f acc[4] = {};
  for (int k0 = 0; k0 < DIMD; k0 += 32) {
    __syncthreads();
    {
      const int e0 = tid * 4;
      const int r  = e0 >> 5;
      const int kk = e0 & 31;
      const f16* src = &ctx[(size_t)(row0 + r) * DIMD + k0 + kk];
      if (k0 + 32 < DIMD) __builtin_prefetch(src + 32, 0, 3);
      As[r][kk + 0] = src[0]; As[r][kk + 1] = src[1];
      As[r][kk + 2] = src[2]; As[r][kk + 3] = src[3];
    }
#pragma unroll
    for (int j = 0; j < 2; ++j) {
      const int n = lane + j * 32;
      const float* src = &out_w[(size_t)(c0 + n) * DIMD + k0];
#pragma unroll
      for (int q4 = 0; q4 < 8; ++q4) {
        float4 w4 = *(const float4*)(src + q4 * 4);
        Bs[wave][n][q4 * 4 + 0] = (f16)w4.x; Bs[wave][n][q4 * 4 + 1] = (f16)w4.y;
        Bs[wave][n][q4 * 4 + 2] = (f16)w4.z; Bs[wave][n][q4 * 4 + 3] = (f16)w4.w;
      }
    }
    __syncthreads();
    v16h a = load_a_frag_lds(&As[0][0], 36, lane);
#pragma unroll
    for (int nt = 0; nt < 4; ++nt) {
      v16h bfr = load_b_frag_lds(&Bs[wave][nt * 16][0], 34, lane);
      acc[nt] = wmma_f16(a, bfr, acc[nt]);
    }
  }
  const int hi = lane >> 4, nl = lane & 15;
#pragma unroll
  for (int nt = 0; nt < 4; ++nt) {
    const float bias = out_b[c0 + nt * 16 + nl];
#pragma unroll
    for (int r = 0; r < 8; ++r) {
      const int m = hi * 8 + r;
      const int c = c0 + nt * 16 + nl;
      out[(size_t)(row0 + m) * DIMD + c] = acc[nt][r] + bias;
    }
  }
}

// ---------------------------------------------------------------------------
extern "C" void kernel_launch(void* const* d_in, const int* in_sizes, int n_in,
                              void* d_out, int out_size, void* d_ws, size_t ws_size,
                              hipStream_t stream) {
  (void)in_sizes; (void)n_in; (void)out_size; (void)ws_size;
  const float* stok  = (const float*)d_in[0];
  const float* stm   = (const float*)d_in[1];
  const unsigned char* stok_mask = (const unsigned char*)d_in[2];
  const unsigned char* stm_mask  = (const unsigned char*)d_in[3];
  const float* qkv_w = (const float*)d_in[4];
  const float* qkv_b = (const float*)d_in[5];
  const float* out_w = (const float*)d_in[6];
  const float* out_b = (const float*)d_in[7];
  const float* rel_emb_k = (const float*)d_in[8];
  const float* rel_emb_v = (const float*)d_in[9];
  const float* stm_emb_k = (const float*)d_in[10];
  const float* stm_emb_v = (const float*)d_in[11];

  // workspace: q,k (head-major), vT (transposed), ctx — all f16, 4 * 10.5 MB
  const size_t perT = (size_t)BATCH * NHEADS * SEQ * HD;
  f16* qb  = (f16*)d_ws;
  f16* kb  = qb + perT;
  f16* vT  = kb + perT;
  f16* ctx = vT + perT;

  // d_out = [final_output (B*S*DIM f32) ; attn (B*H*S*S f32)]
  float* final_out = (float*)d_out;
  float* attn_out  = final_out + (size_t)BATCH * SEQ * DIMD;

  qkv_proj_kernel<<<dim3(BATCH * SEQ / 16, (3 * DIMD) / 256), 128, 0, stream>>>(
      stok, stm, qkv_w, qkv_b, qb, kb, vT);

  attn_kernel<<<dim3(BATCH * NHEADS * (SEQ / 16)), 32, 0, stream>>>(
      qb, kb, vT, stok_mask, stm_mask, rel_emb_k, rel_emb_v, stm_emb_k, stm_emb_v,
      attn_out, ctx);

  out_proj_kernel<<<dim3(BATCH * SEQ / 16, DIMD / 256), 128, 0, stream>>>(
      ctx, out_w, out_b, final_out);
}
